// Model_78752520340026
// MI455X (gfx1250) — matrix operand order repaired and checked by reading, not compiled, not run on hardware
//
#include <hip/hip_runtime.h>
#include <math.h>

// ---------------- problem constants ----------------
#define Bn   16
#define Nn   4096
#define Cc   64
#define Ss   1024
#define Kk   24
#define OUTC 128
#define DINx 131   // 2C+3
#define CP3  67    // C+3
#define KPAD 160   // DIN padded to multiple of 32
#define EPSF 1e-5f

typedef __attribute__((ext_vector_type(16))) _Float16 v16h;
typedef __attribute__((ext_vector_type(8)))  _Float16 v8h;
typedef __attribute__((ext_vector_type(4)))  _Float16 v4h;
typedef __attribute__((ext_vector_type(8)))  float    v8f;

// ---------------- WMMA fragment helpers (contiguous k order, wave32) ----------------
// A: 16x32 f16 tile from row-major weight matrix (lda elements).
// a[0..15] = K kb..kb+15; kb = k0 (+16 for lanes 16-31).
// Rows contiguous -> two 16B vector loads.
__device__ __forceinline__ v16h load_frag_a(const _Float16* A, int lda, int m0, int k0, int lane) {
  const _Float16* row = A + (size_t)(m0 + (lane & 15)) * lda;
  int kb = k0 + ((lane & 16) ? 16 : 0);
  v8h lo = *(const v8h*)(row + kb);
  v8h hi = *(const v8h*)(row + kb + 8);
  v16h a;
#pragma unroll
  for (int r = 0; r < 8; ++r) { a[r] = lo[r]; a[8 + r] = hi[r]; }
  return a;
}

// B: 32x16 f16 tile from a COLUMN-MAJOR matrix Bc[col][K] (ld elements per column).
// b[i] = K = kb+i for this lane's column; contiguous 32B -> vector load.
// lanes 0-15: col n0+lane, kb=k0; lanes 16-31: col n0+lane-16, kb=k0+16.
__device__ __forceinline__ v16h load_frag_b_cm(const _Float16* Bc, int ld, int k0, int n0, int lane) {
  const _Float16* p = Bc + (size_t)(n0 + (lane & 15)) * ld + k0 + ((lane & 16) ? 16 : 0);
  return *(const v16h*)p;
}

__device__ __forceinline__ v8f wmma_f16(v16h a, v16h b, v8f c) {
  return __builtin_amdgcn_wmma_f32_16x16x32_f16(false, a, false, b, (short)0, c, false, false);
}

// ---------------- 1) fold BN into weights, convert to f16, zero accumulators ----------------
__global__ void prep_weights_kernel(
    const float* Wt, const float* bt, const float* st, const float* ct,
    const float* W1, const float* b1, const float* s1, const float* c1,
    const float* W2, const float* b2, const float* s2, const float* c2,
    const float* W3, const float* b3, const float* s3, const float* c3,
    const float* W4, const float* b4, const float* s4, const float* c4,
    _Float16* Wt_h, float* bt_f, _Float16* W1_h, float* b1f,
    _Float16* W2_h, float* b2f, _Float16* W3_h, float* b3f,
    _Float16* W4_h, float* b4f, float* accb) {
  int t = blockIdx.x * blockDim.x + threadIdx.x;
  int stride = gridDim.x * blockDim.x;
  for (int e = t; e < OUTC * KPAD; e += stride) {
    int o = e / KPAD, i = e - o * KPAD;
    float v = (i < DINx) ? st[o] * Wt[o * DINx + i] : 0.f;
    Wt_h[e] = (_Float16)v;
  }
  for (int e = t; e < OUTC * OUTC; e += stride) {
    int o = e >> 7;
    W1_h[e] = (_Float16)(s1[o] * W1[e]);
    W2_h[e] = (_Float16)(s2[o] * W2[e]);
    W3_h[e] = (_Float16)(s3[o] * W3[e]);
    W4_h[e] = (_Float16)(s4[o] * W4[e]);
  }
  for (int o = t; o < OUTC; o += stride) {
    bt_f[o] = st[o] * bt[o] + ct[o];
    b1f[o]  = s1[o] * b1[o] + c1[o];
    b2f[o]  = s2[o] * b2[o] + c2[o];
    b3f[o]  = s3[o] * b3[o] + c3[o];
    b4f[o]  = s4[o] * b4[o] + c4[o];
  }
  if (t < Bn) accb[t] = 0.f;
}

// ---------------- 2) furthest point sampling: one workgroup per batch ----------------
// xyz cached in LDS, per-thread dist in registers, wave32 shfl reduction (3 barriers/step).
#define FPS_T 256
__global__ void __launch_bounds__(FPS_T) fps_kernel(const float* xyz, int* fps_idx) {
  __shared__ float sx[Nn], sy[Nn], sz[Nn];   // 48 KB of the 320 KB WGP LDS
  __shared__ float rvalW[FPS_T / 32];
  __shared__ int   ridxW[FPS_T / 32];
  int b = blockIdx.x, t = threadIdx.x;
  int lane = t & 31, wv = t >> 5;
  const float* base = xyz + (size_t)b * Nn * 3;
  float dcur[Nn / FPS_T];  // register resident
#pragma unroll
  for (int j = 0; j < Nn / FPS_T; ++j) {
    int i = t + j * FPS_T;
    sx[i] = base[i * 3 + 0];
    sy[i] = base[i * 3 + 1];
    sz[i] = base[i * 3 + 2];
    dcur[j] = 1e10f;
  }
  if (t == 0) fps_idx[b * Ss] = 0;
  __syncthreads();
  int last = 0;
  for (int step = 1; step < Ss; ++step) {
    float px = sx[last], py = sy[last], pz = sz[last];
    float bv = -1.f; int bi = 0;
#pragma unroll
    for (int j = 0; j < Nn / FPS_T; ++j) {
      int i = t + j * FPS_T;
      float dx = sx[i] - px, dy = sy[i] - py, dz = sz[i] - pz;
      float d  = dx * dx + dy * dy + dz * dz;
      float nd = fminf(dcur[j], d);
      dcur[j] = nd;
      if (nd > bv) { bv = nd; bi = i; }   // ascending i: strict > keeps first index
    }
    // wave32 argmax reduction (prefer smaller index on ties)
#pragma unroll
    for (int off = 16; off > 0; off >>= 1) {
      float v2 = __shfl_down(bv, off, 32);
      int   i2 = __shfl_down(bi, off, 32);
      if (v2 > bv || (v2 == bv && i2 < bi)) { bv = v2; bi = i2; }
    }
    if (lane == 0) { rvalW[wv] = bv; ridxW[wv] = bi; }
    __syncthreads();
    if (t == 0) {
      float best = rvalW[0]; int bidx = ridxW[0];
#pragma unroll
      for (int w = 1; w < FPS_T / 32; ++w) {
        float v2 = rvalW[w]; int i2 = ridxW[w];
        if (v2 > best || (v2 == best && i2 < bidx)) { best = v2; bidx = i2; }
      }
      ridxW[0] = bidx;
      fps_idx[b * Ss + step] = bidx;
    }
    __syncthreads();
    last = ridxW[0];
    __syncthreads();
  }
}

// ---------------- 3) gather centroids ----------------
__global__ void gather_kernel(const float* xyz, const float* points, const int* fps_idx,
                              float* new_xyz, float* new_pts) {
  int bs = blockIdx.x * blockDim.x + threadIdx.x;
  if (bs >= Bn * Ss) return;
  int b = bs >> 10;
  int pi = fps_idx[bs];
  const float* sp3 = xyz + ((size_t)b * Nn + pi) * 3;
  new_xyz[bs * 3 + 0] = sp3[0];
  new_xyz[bs * 3 + 1] = sp3[1];
  new_xyz[bs * 3 + 2] = sp3[2];
  const float* sp = points + ((size_t)b * Nn + pi) * Cc;
  float* dp = new_pts + (size_t)bs * Cc;
  for (int c = 0; c < Cc; ++c) dp[c] = sp[c];
}

// ---------------- 4) kNN (top-24 smallest d2), one thread per query ----------------
#define KNN_BLOCK 128
#define KNN_TILE  512
__global__ void __launch_bounds__(KNN_BLOCK) knn_kernel(const float* xyz, const float* new_xyz,
                                                        int* knn_idx) {
  __shared__ float tx[KNN_TILE], ty[KNN_TILE], tz[KNN_TILE];
  int b = blockIdx.y;
  int s = blockIdx.x * KNN_BLOCK + threadIdx.x;
  int bs = b * Ss + s;
  float qx = new_xyz[bs * 3 + 0], qy = new_xyz[bs * 3 + 1], qz = new_xyz[bs * 3 + 2];
  float bd[Kk]; int bi[Kk];
#pragma unroll
  for (int j = 0; j < Kk; ++j) { bd[j] = 3.4e38f; bi[j] = 0; }
  float worst = 3.4e38f; int wslot = 0;
  for (int t0 = 0; t0 < Nn; t0 += KNN_TILE) {
    __syncthreads();
    for (int i = threadIdx.x; i < KNN_TILE; i += KNN_BLOCK) {
      const float* p = xyz + ((size_t)b * Nn + t0 + i) * 3;
      tx[i] = p[0]; ty[i] = p[1]; tz[i] = p[2];
    }
    __syncthreads();
    for (int i = 0; i < KNN_TILE; ++i) {
      float dx = tx[i] - qx, dy = ty[i] - qy, dz = tz[i] - qz;
      float d = dx * dx + dy * dy + dz * dz;
      if (d < worst) {
#pragma unroll
        for (int j = 0; j < Kk; ++j) if (j == wslot) { bd[j] = d; bi[j] = t0 + i; }
        worst = -1.f;
#pragma unroll
        for (int j = 0; j < Kk; ++j) if (bd[j] > worst) { worst = bd[j]; wslot = j; }
      }
    }
  }
  int* outp = knn_idx + (size_t)bs * Kk;
#pragma unroll
  for (int j = 0; j < Kk; ++j) outp[j] = bi[j];
}

// ---------------- 5) group stats: per-(b,s,ch) mean + batch sum of centered^2 ----------------
__global__ void __launch_bounds__(128) stats_kernel(const float* xyz, const float* points,
                                                    const int* knn_idx, float* mean, float* accb) {
  __shared__ float part[128];
  int bs = blockIdx.x;
  int b  = bs >> 10;
  int ch = threadIdx.x;
  const int* idxp = knn_idx + (size_t)bs * Kk;
  float p = 0.f;
  if (ch < CP3) {
    float sum = 0.f, sq = 0.f;
    for (int k = 0; k < Kk; ++k) {
      int pi = idxp[k];
      float v = (ch < Cc) ? points[((size_t)b * Nn + pi) * Cc + ch]
                          : xyz[((size_t)b * Nn + pi) * 3 + (ch - Cc)];
      sum += v; sq += v * v;
    }
    float m = sum * (1.f / (float)Kk);
    mean[(size_t)bs * CP3 + ch] = m;
    p = sq - (float)Kk * m * m;   // sum_k (v-m)^2
  }
  part[ch] = p;
  __syncthreads();
  for (int off = 64; off > 0; off >>= 1) {
    if (ch < off) part[ch] += part[ch + off];
    __syncthreads();
  }
  if (ch == 0) atomicAdd(&accb[b], part[0]);
}

__global__ void finalize_kernel(const float* accb, float* stdv) {
  int b = threadIdx.x;
  if (b < Bn) stdv[b] = sqrtf(accb[b] / (float)(Ss * Kk * CP3 - 1));  // ddof=1, mean(centered)==0
}

// ---------------- 6) PreExtraction: column-major LDS tiles, 3 WMMA convs, max-pool over K --------
// Activation tiles stored column-major [col][K] so B-fragments are one contiguous 32B LDS load and
// D writebacks (8 consecutive out channels per v8f) are one 16B LDS store.
__global__ void __launch_bounds__(64) pre_extract_kernel(
    const float* xyz, const float* points, const float* alpha, const float* beta,
    const float* new_pts, const int* knn_idx, const float* mean, const float* stdv,
    const _Float16* Wt_h, const float* bt_f,
    const _Float16* W1_h, const float* b1f,
    const _Float16* W2_h, const float* b2f,
    float* pre_f32, _Float16* pre_f16T) {
  __shared__ _Float16 X0c[32 * KPAD];   // [k][ch]   10 KB
  __shared__ _Float16 X1c[32 * OUTC];   // [k][o]     8 KB
  __shared__ _Float16 X2c[32 * OUTC];   //            8 KB
  __shared__ float    REDc[32 * OUTC];  // [k][o]    16 KB
  int bs = blockIdx.x;
  int b = bs >> 10, s = bs & 1023;
  int tid = threadIdx.x;
  const int* idxp = knn_idx + (size_t)bs * Kk;
  float sinv = 1.f / (stdv[b] + EPSF);

  // build normalized grouped features + anchors, column-major [k][ch], 4-ch vector chunks
  for (int e = tid; e < 32 * (KPAD / 4); e += 64) {
    int k  = e / (KPAD / 4);
    int c4 = (e - k * (KPAD / 4)) * 4;
    v4h pk;
    if (k < Kk) {
      int pi = idxp[k];
#pragma unroll
      for (int u = 0; u < 4; ++u) {
        int ch = c4 + u;
        float x = 0.f;
        if (ch < CP3) {
          float raw = (ch < Cc) ? points[((size_t)b * Nn + pi) * Cc + ch]
                                : xyz[((size_t)b * Nn + pi) * 3 + (ch - Cc)];
          x = alpha[ch] * ((raw - mean[(size_t)bs * CP3 + ch]) * sinv) + beta[ch];
        } else if (ch < DINx) {
          x = new_pts[(size_t)bs * Cc + (ch - CP3)];   // repeated anchor feats
        }
        pk[u] = (_Float16)x;
      }
    } else {
#pragma unroll
      for (int u = 0; u < 4; ++u) pk[u] = (_Float16)0.f;
    }
    *(v4h*)&X0c[(size_t)k * KPAD + c4] = pk;
  }
  __syncthreads();

  int lane  = tid & 31;
  int n0    = (tid >> 5) * 16;         // wave's N-tile (columns = neighbors)
  int col   = lane & 15;
  int rbase = (lane & 16) ? 8 : 0;
  int gc    = n0 + col;                // this lane's global column
  v8f zacc = {};
  v8f acc[8];

  // ---- layer0: 128x160 @ 160x32 ----
#pragma unroll
  for (int mt = 0; mt < 8; ++mt) acc[mt] = zacc;
#pragma unroll
  for (int kt = 0; kt < 5; ++kt) {
    v16h bf = load_frag_b_cm(X0c, KPAD, kt * 32, n0, lane);
#pragma unroll
    for (int mt = 0; mt < 8; ++mt) {
      v16h af = load_frag_a(Wt_h, KPAD, mt * 16, kt * 32, lane);
      acc[mt] = wmma_f16(af, bf, acc[mt]);
    }
  }
#pragma unroll
  for (int mt = 0; mt < 8; ++mt) {
    int o0 = mt * 16 + rbase;
    v8h pk;
#pragma unroll
    for (int r = 0; r < 8; ++r) pk[r] = (_Float16)fmaxf(acc[mt][r] + bt_f[o0 + r], 0.f);
    *(v8h*)&X1c[(size_t)gc * OUTC + o0] = pk;
  }
  __syncthreads();

  // ---- layer1: relu(W1 @ X1 + b1) -> X2 ----
#pragma unroll
  for (int mt = 0; mt < 8; ++mt) acc[mt] = zacc;
#pragma unroll
  for (int kt = 0; kt < 4; ++kt) {
    v16h bf = load_frag_b_cm(X1c, OUTC, kt * 32, n0, lane);
#pragma unroll
    for (int mt = 0; mt < 8; ++mt) {
      v16h af = load_frag_a(W1_h, OUTC, mt * 16, kt * 32, lane);
      acc[mt] = wmma_f16(af, bf, acc[mt]);
    }
  }
#pragma unroll
  for (int mt = 0; mt < 8; ++mt) {
    int o0 = mt * 16 + rbase;
    v8h pk;
#pragma unroll
    for (int r = 0; r < 8; ++r) pk[r] = (_Float16)fmaxf(acc[mt][r] + b1f[o0 + r], 0.f);
    *(v8h*)&X2c[(size_t)gc * OUTC + o0] = pk;
  }
  __syncthreads();

  // ---- layer2: relu(W2 @ X2 + b2 + X1) ----
#pragma unroll
  for (int mt = 0; mt < 8; ++mt) acc[mt] = zacc;
#pragma unroll
  for (int kt = 0; kt < 4; ++kt) {
    v16h bf = load_frag_b_cm(X2c, OUTC, kt * 32, n0, lane);
#pragma unroll
    for (int mt = 0; mt < 8; ++mt) {
      v16h af = load_frag_a(W2_h, OUTC, mt * 16, kt * 32, lane);
      acc[mt] = wmma_f16(af, bf, acc[mt]);
    }
  }
#pragma unroll
  for (int mt = 0; mt < 8; ++mt) {
    int o0 = mt * 16 + rbase;
    v8h xres = *(const v8h*)&X1c[(size_t)gc * OUTC + o0];
    v8f ov;
#pragma unroll
    for (int r = 0; r < 8; ++r)
      ov[r] = fmaxf(acc[mt][r] + b2f[o0 + r] + (float)xres[r], 0.f);
    *(v8f*)&REDc[(size_t)gc * OUTC + o0] = ov;  // pad cols never read by the pool
  }
  __syncthreads();

  // ---- max-pool over the K=24 valid neighbors ----
  for (int o = tid; o < OUTC; o += 64) {
    float m = REDc[o];
    for (int j = 1; j < Kk; ++j) m = fmaxf(m, REDc[(size_t)j * OUTC + o]);
    pre_f32[((size_t)b * OUTC + o) * Ss + s] = m;                 // [B][128][S] for residual
    pre_f16T[(size_t)bs * OUTC + o]          = (_Float16)m;       // [B][S][128] for WMMA B loads
  }
}

// ---------------- 7) PosExtraction res-block over [128 x S] ----------------
__global__ void __launch_bounds__(64) pos_extract_kernel(
    const _Float16* pre_f16T, const float* pre_f32,
    const _Float16* W3_h, const float* b3f,
    const _Float16* W4_h, const float* b4f,
    float* out) {
  __shared__ _Float16 A1c[32 * OUTC];   // [colLocal][o], column-major
  int b = blockIdx.y;
  int scol0 = blockIdx.x * 32;
  int tid = threadIdx.x, lane = tid & 31;
  int n0 = (tid >> 5) * 16;
  int col = lane & 15;
  int rbase = (lane & 16) ? 8 : 0;
  const _Float16* XcT = pre_f16T + (size_t)b * Ss * OUTC;   // column-major [s][128]
  v8f zacc = {};
  v8f acc[8];

  // conv3: relu(W3 @ X + b3) -> A1 (LDS)
#pragma unroll
  for (int mt = 0; mt < 8; ++mt) acc[mt] = zacc;
#pragma unroll
  for (int kt = 0; kt < 4; ++kt) {
    v16h bf = load_frag_b_cm(XcT, OUTC, kt * 32, scol0 + n0, lane);
#pragma unroll
    for (int mt = 0; mt < 8; ++mt) {
      v16h af = load_frag_a(W3_h, OUTC, mt * 16, kt * 32, lane);
      acc[mt] = wmma_f16(af, bf, acc[mt]);
    }
  }
#pragma unroll
  for (int mt = 0; mt < 8; ++mt) {
    int o0 = mt * 16 + rbase;
    v8h pk;
#pragma unroll
    for (int r = 0; r < 8; ++r) pk[r] = (_Float16)fmaxf(acc[mt][r] + b3f[o0 + r], 0.f);
    *(v8h*)&A1c[(size_t)(n0 + col) * OUTC + o0] = pk;
  }
  __syncthreads();

  // conv4 + residual + relu -> out
#pragma unroll
  for (int mt = 0; mt < 8; ++mt) acc[mt] = zacc;
#pragma unroll
  for (int kt = 0; kt < 4; ++kt) {
    v16h bf = load_frag_b_cm(A1c, OUTC, kt * 32, n0, lane);
#pragma unroll
    for (int mt = 0; mt < 8; ++mt) {
      v16h af = load_frag_a(W4_h, OUTC, mt * 16, kt * 32, lane);
      acc[mt] = wmma_f16(af, bf, acc[mt]);
    }
  }
  int sc = scol0 + n0 + col;
#pragma unroll
  for (int mt = 0; mt < 8; ++mt)
#pragma unroll
    for (int r = 0; r < 8; ++r) {
      int o = mt * 16 + rbase + r;
      float y = acc[mt][r] + b4f[o] + pre_f32[((size_t)b * OUTC + o) * Ss + sc];
      out[((size_t)b * OUTC + o) * Ss + sc] = fmaxf(y, 0.f);
    }
}

// ---------------- host launcher ----------------
extern "C" void kernel_launch(void* const* d_in, const int* in_sizes, int n_in,
                              void* d_out, int out_size, void* d_ws, size_t ws_size,
                              hipStream_t stream) {
  (void)in_sizes; (void)n_in; (void)out_size; (void)ws_size;
  const float* xyz    = (const float*)d_in[0];
  const float* points = (const float*)d_in[1];
  const float* alpha  = (const float*)d_in[2];
  const float* beta   = (const float*)d_in[3];
  const float* Wt = (const float*)d_in[4];  const float* bt = (const float*)d_in[5];
  const float* st = (const float*)d_in[6];  const float* ct = (const float*)d_in[7];
  const float* W1 = (const float*)d_in[8];  const float* b1 = (const float*)d_in[9];
  const float* s1 = (const float*)d_in[10]; const float* c1 = (const float*)d_in[11];
  const float* W2 = (const float*)d_in[12]; const float* b2 = (const float*)d_in[13];
  const float* s2 = (const float*)d_in[14]; const float* c2 = (const float*)d_in[15];
  const float* W3 = (const float*)d_in[16]; const float* b3 = (const float*)d_in[17];
  const float* s3 = (const float*)d_in[18]; const float* c3 = (const float*)d_in[19];
  const float* W4 = (const float*)d_in[20]; const float* b4 = (const float*)d_in[21];
  const float* s4 = (const float*)d_in[22]; const float* c4 = (const float*)d_in[23];
  float* out = (float*)d_out;

  char* ws = (char*)d_ws;
  size_t off = 0;
  auto carve = [&](size_t bytes) -> void* {
    void* p = (void*)(ws + off);
    off += (bytes + 255) & ~(size_t)255;
    return p;
  };
  int*      fps_idx  = (int*)      carve((size_t)Bn * Ss * 4);
  float*    new_xyz  = (float*)    carve((size_t)Bn * Ss * 3 * 4);
  float*    new_pts  = (float*)    carve((size_t)Bn * Ss * Cc * 4);
  int*      knn_idx  = (int*)      carve((size_t)Bn * Ss * Kk * 4);
  float*    meanb    = (float*)    carve((size_t)Bn * Ss * CP3 * 4);
  float*    accb     = (float*)    carve((size_t)Bn * 4);
  float*    stdv     = (float*)    carve((size_t)Bn * 4);
  _Float16* Wt_h     = (_Float16*) carve((size_t)OUTC * KPAD * 2);
  _Float16* W1_h     = (_Float16*) carve((size_t)OUTC * OUTC * 2);
  _Float16* W2_h     = (_Float16*) carve((size_t)OUTC * OUTC * 2);
  _Float16* W3_h     = (_Float16*) carve((size_t)OUTC * OUTC * 2);
  _Float16* W4_h     = (_Float16*) carve((size_t)OUTC * OUTC * 2);
  float*    bt_f     = (float*)    carve((size_t)OUTC * 4);
  float*    b1f      = (float*)    carve((size_t)OUTC * 4);
  float*    b2f      = (float*)    carve((size_t)OUTC * 4);
  float*    b3f      = (float*)    carve((size_t)OUTC * 4);
  float*    b4f      = (float*)    carve((size_t)OUTC * 4);
  float*    pre_f32  = (float*)    carve((size_t)Bn * OUTC * Ss * 4);
  _Float16* pre_f16T = (_Float16*) carve((size_t)Bn * Ss * OUTC * 2);

  prep_weights_kernel<<<64, 256, 0, stream>>>(
      Wt, bt, st, ct, W1, b1, s1, c1, W2, b2, s2, c2,
      W3, b3, s3, c3, W4, b4, s4, c4,
      Wt_h, bt_f, W1_h, b1f, W2_h, b2f, W3_h, b3f, W4_h, b4f, accb);

  fps_kernel<<<Bn, FPS_T, 0, stream>>>(xyz, fps_idx);

  gather_kernel<<<(Bn * Ss + 255) / 256, 256, 0, stream>>>(xyz, points, fps_idx, new_xyz, new_pts);

  knn_kernel<<<dim3(Ss / KNN_BLOCK, Bn), KNN_BLOCK, 0, stream>>>(xyz, new_xyz, knn_idx);

  stats_kernel<<<Bn * Ss, 128, 0, stream>>>(xyz, points, knn_idx, meanb, accb);

  finalize_kernel<<<1, 32, 0, stream>>>(accb, stdv);

  pre_extract_kernel<<<Bn * Ss, 64, 0, stream>>>(
      xyz, points, alpha, beta, new_pts, knn_idx, meanb, stdv,
      Wt_h, bt_f, W1_h, b1f, W2_h, b2f, pre_f32, pre_f16T);

  pos_extract_kernel<<<dim3(Ss / 32, Bn), 64, 0, stream>>>(
      pre_f16T, pre_f32, W3_h, b3f, W4_h, b4f, out);
}
